// Encoder_50525995270410
// MI455X (gfx1250) — matrix operand order repaired and checked by reading, not compiled, not run on hardware
//
#include <hip/hip_runtime.h>
#include <hip/hip_bf16.h>
#include <stdint.h>

typedef __attribute__((ext_vector_type(16))) __bf16 bf16x16;
typedef __attribute__((ext_vector_type(8)))  float  f32x8;

#define CH 128          // IN_CH == HID == 128
#define LAT 64

static __device__ __forceinline__ unsigned short f32_to_bf16_rne(float f) {
    unsigned int u = __float_as_uint(f);
    unsigned int lsb = (u >> 16) & 1u;
    u += 0x7fffu + lsb;
    return (unsigned short)(u >> 16);
}

// ---------------- degree / normalization ----------------
__global__ void k_deg_init(float* deg, int n) {
    int i = blockIdx.x * blockDim.x + threadIdx.x;
    if (i < n) deg[i] = 1.0f;                       // self loop
}
__global__ void k_deg_accum(const int* __restrict__ dst, float* deg, int e) {
    int i = blockIdx.x * blockDim.x + threadIdx.x;
    if (i < e) atomicAdd(&deg[dst[i]], 1.0f);
}
__global__ void k_dinv(float* deg, int n) {
    int i = blockIdx.x * blockDim.x + threadIdx.x;
    if (i < n) { float d = deg[i]; deg[i] = (d > 0.0f) ? rsqrtf(d) : 0.0f; }
}

// ---------------- conversions / repack ----------------
__global__ void k_cvt_bf16(const float* __restrict__ X, unsigned short* __restrict__ Y, int n) {
    int i = blockIdx.x * blockDim.x + threadIdx.x;
    if (i < n) Y[i] = f32_to_bf16_rne(X[i]);
}

// Pack W[128][128] (row-major K x N, f32) into WMMA B-fragment order (bf16):
// dst[ ((kk*8+nt)*32 + lane)*16 + e ], per the ISA 16-bit B layout:
//   n = nt*16 + (lane&15); kbase = kk*32 + (lane>>4)*8; k = kbase + (e<8 ? e : e+8)
__global__ void k_repack_w(const float* __restrict__ W, unsigned short* __restrict__ Bf) {
    int t = blockIdx.x * blockDim.x + threadIdx.x;
    if (t >= 4 * 8 * 32 * 16) return;
    int e    = t & 15;
    int lane = (t >> 4) & 31;
    int nt   = (t >> 9) & 7;
    int kk   = t >> 12;
    int n = nt * 16 + (lane & 15);
    int k = kk * 32 + ((lane >> 4) * 8) + (e < 8 ? e : e + 8);
    Bf[t] = f32_to_bf16_rne(W[k * CH + n]);
}

// ---------------- WMMA GEMM: H[M][128] = Ab[M][128] @ Wfrag ----------------
__global__ __launch_bounds__(256) void k_gemm_bf16(
    const unsigned short* __restrict__ Ab,   // [M][128] bf16
    const unsigned short* __restrict__ Wf,   // fragment-packed, 16384 bf16
    float* __restrict__ H,                   // [M][128] f32
    int M, int mtiles)
{
    __shared__ unsigned short Wlds[16384];   // 32 KB, shared by 8 waves
    {   // cooperative stage: 256 threads x 8 x 16B = 32 KB
        const uint4* g = reinterpret_cast<const uint4*>(Wf);
        uint4*       l = reinterpret_cast<uint4*>(Wlds);
#pragma unroll
        for (int i = 0; i < 8; ++i)
            l[threadIdx.x + 256 * i] = g[threadIdx.x + 256 * i];
    }
    __syncthreads();

    int wave = blockIdx.x * (blockDim.x >> 5) + (threadIdx.x >> 5);
    if (wave >= mtiles) return;
    int lane  = threadIdx.x & 31;
    int row0  = wave * 16;
    int m     = row0 + (lane & 15);
    if (m >= M) m = M - 1;                   // clamp (keeps EXEC all-ones)
    int khalf = (lane >> 4) * 8;             // A layout K-half per lane group

    f32x8 acc[8] = {};
    const unsigned short* arow = Ab + (size_t)m * CH;

#pragma unroll
    for (int kk = 0; kk < 4; ++kk) {
        bf16x16 a;
        uint4* ap = reinterpret_cast<uint4*>(&a);
        // 16-bit A 16x32 layout: elems 0..7 = K[khalf..khalf+7], 8..15 = K[khalf+16..khalf+23]
        ap[0] = *reinterpret_cast<const uint4*>(arow + kk * 32 + khalf);
        ap[1] = *reinterpret_cast<const uint4*>(arow + kk * 32 + khalf + 16);
#pragma unroll
        for (int nt = 0; nt < 8; ++nt) {
            bf16x16 b = *reinterpret_cast<const bf16x16*>(Wlds + (((kk * 8 + nt) * 32 + lane) << 4));
            acc[nt] = __builtin_amdgcn_wmma_f32_16x16x32_bf16(
                false, a, false, b, (short)0, acc[nt], false, false);
        }
    }

    // D layout: VGPR v -> row (v + 8*(lane>=16)), col lane&15
    int mbase = row0 + ((lane >> 4) * 8);
    int ncol  = lane & 15;
#pragma unroll
    for (int nt = 0; nt < 8; ++nt)
#pragma unroll
        for (int v = 0; v < 8; ++v) {
            int row = mbase + v;
            if (row < M) H[(size_t)row * CH + nt * 16 + ncol] = acc[nt][v];
        }
}

// ---------------- aggregation ----------------
// one wave per node; wave-uniform node index -> scalar-addressed dinv load
__global__ void k_selfloop_init(const float* __restrict__ H, const float* __restrict__ dinv,
                                float* __restrict__ OUT, int n) {
    int wave = blockIdx.x * (blockDim.x >> 5) + (threadIdx.x >> 5);
    int i = __builtin_amdgcn_readfirstlane(wave);
    if (i >= n) return;
    int lane = threadIdx.x & 31;
    float di = dinv[i], s = di * di;
    float4 h = reinterpret_cast<const float4*>(H + (size_t)i * CH)[lane];
    float4 o; o.x = h.x * s; o.y = h.y * s; o.z = h.z * s; o.w = h.w * s;
    reinterpret_cast<float4*>(OUT + (size_t)i * CH)[lane] = o;
}

// one wave per edge; src/dst/norm forced into SGPRs so the gather/atomic
// addressing is saddr+voffset and per-edge metadata loads are uniform
__global__ void k_edge_scatter(const int* __restrict__ src, const int* __restrict__ dst,
                               const float* __restrict__ dinv, const float* __restrict__ H,
                               float* __restrict__ OUT, int e) {
    int wave = blockIdx.x * (blockDim.x >> 5) + (threadIdx.x >> 5);
    int ed = __builtin_amdgcn_readfirstlane(wave);
    if (ed >= e) return;
    int lane = threadIdx.x & 31;
    int s = __builtin_amdgcn_readfirstlane(src[ed]);
    int d = __builtin_amdgcn_readfirstlane(dst[ed]);
    float norm = dinv[s] * dinv[d];
    float4 h = reinterpret_cast<const float4*>(H + (size_t)s * CH)[lane];
    float* ob = OUT + (size_t)d * CH + lane * 4;
    atomicAdd(ob + 0, h.x * norm);
    atomicAdd(ob + 1, h.y * norm);
    atomicAdd(ob + 2, h.z * norm);
    atomicAdd(ob + 3, h.w * norm);
}

__global__ void k_bias_relu_bf16(const float* __restrict__ OUT, const float* __restrict__ bias,
                                 unsigned short* __restrict__ Ab, int n) {
    int t = blockIdx.x * blockDim.x + threadIdx.x;
    if (t >= n * CH) return;
    float v = OUT[t] + bias[t & (CH - 1)];
    Ab[t] = f32_to_bf16_rne(v > 0.0f ? v : 0.0f);
}
__global__ void k_bias_relu_f32(float* OUT, const float* __restrict__ bias, int n) {
    int t = blockIdx.x * blockDim.x + threadIdx.x;
    if (t >= n * CH) return;
    float v = OUT[t] + bias[t & (CH - 1)];
    OUT[t] = v > 0.0f ? v : 0.0f;
}

// ---------------- pooling + heads ----------------
__global__ void k_pool_zero(float* gsum, float* gcnt, int G) {
    int t = blockIdx.x * blockDim.x + threadIdx.x;
    if (t < G * CH) gsum[t] = 0.0f;
    if (t < G) gcnt[t] = 0.0f;
}
__global__ void k_pool_cnt(const int* __restrict__ batch, float* gcnt, int n) {
    int i = blockIdx.x * blockDim.x + threadIdx.x;
    if (i < n) atomicAdd(&gcnt[batch[i]], 1.0f);
}
__global__ void k_pool_sum(const int* __restrict__ batch, const float* __restrict__ Hf,
                           float* gsum, int n) {
    int wave = blockIdx.x * (blockDim.x >> 5) + (threadIdx.x >> 5);
    int i = __builtin_amdgcn_readfirstlane(wave);
    if (i >= n) return;
    int lane = threadIdx.x & 31;
    int g = __builtin_amdgcn_readfirstlane(batch[i]);
    float4 h = reinterpret_cast<const float4*>(Hf + (size_t)i * CH)[lane];
    float* ob = gsum + (size_t)g * CH + lane * 4;
    atomicAdd(ob + 0, h.x);
    atomicAdd(ob + 1, h.y);
    atomicAdd(ob + 2, h.z);
    atomicAdd(ob + 3, h.w);
}
__global__ void k_gmean(float* gsum, const float* __restrict__ gcnt, int G) {
    int t = blockIdx.x * blockDim.x + threadIdx.x;
    if (t >= G * CH) return;
    gsum[t] /= fmaxf(gcnt[t >> 7], 1.0f);
}
__global__ void k_head(const float* __restrict__ g,
                       const float* __restrict__ Wmu, const float* __restrict__ bmu,
                       const float* __restrict__ Wlv, const float* __restrict__ blv,
                       float* __restrict__ out, int G) {
    int t = blockIdx.x * blockDim.x + threadIdx.x;
    if (t >= G * LAT) return;
    int gi = t >> 6, j = t & (LAT - 1);
    float am = bmu[j], al = blv[j];
    const float* gr = g + (size_t)gi * CH;
#pragma unroll 8
    for (int k = 0; k < CH; ++k) {
        float gv = gr[k];
        am += gv * Wmu[k * LAT + j];
        al += gv * Wlv[k * LAT + j];
    }
    out[(size_t)gi * LAT + j] = am;
    out[(size_t)G * LAT + (size_t)gi * LAT + j] = al;
}

// ---------------- driver ----------------
extern "C" void kernel_launch(void* const* d_in, const int* in_sizes, int n_in,
                              void* d_out, int out_size, void* d_ws, size_t ws_size,
                              hipStream_t stream) {
    const float* x    = (const float*)d_in[0];
    const int*   edge = (const int*)d_in[1];
    const int*   bat  = (const int*)d_in[2];
    const float* W1   = (const float*)d_in[3];
    const float* b1   = (const float*)d_in[4];
    const float* W2   = (const float*)d_in[5];
    const float* b2   = (const float*)d_in[6];
    const float* Wmu  = (const float*)d_in[7];
    const float* bmu  = (const float*)d_in[8];
    const float* Wlv  = (const float*)d_in[9];
    const float* blv  = (const float*)d_in[10];
    float* out = (float*)d_out;

    const int N = in_sizes[0] / CH;          // 100000
    const int E = in_sizes[1] / 2;           // 3200000
    const int G = out_size / (2 * LAT);      // 1024
    const int* src = edge;
    const int* dst = edge + E;

    // workspace carve-out (256B aligned slices)
    char* ws = (char*)d_ws;
    size_t off = 0;
    auto take = [&](size_t bytes) { void* p = ws + off; off += (bytes + 255) & ~(size_t)255; return p; };
    float*          dinv = (float*)take((size_t)N * 4);
    unsigned short* Ab   = (unsigned short*)take((size_t)N * CH * 2);
    unsigned short* Wf1  = (unsigned short*)take(16384 * 2);
    unsigned short* Wf2  = (unsigned short*)take(16384 * 2);
    float*          Hbuf = (float*)take((size_t)N * CH * 4);
    float*          OUT  = (float*)take((size_t)N * CH * 4);
    float*          gsum = (float*)take((size_t)G * CH * 4);
    float*          gcnt = (float*)take((size_t)G * 4);
    (void)ws_size; (void)n_in;

    const int B = 256;                       // 8 waves per block
    const int mtiles = (N + 15) / 16;
    const int gemmBlocks = (mtiles + 7) / 8;
    auto gridN   = (unsigned)((N + B - 1) / B);
    auto gridE   = (unsigned)((E + B - 1) / B);
    auto gridNC  = (unsigned)(((size_t)N * CH + B - 1) / B);
    auto gridNW  = (unsigned)((N + 7) / 8);  // one wave per node
    auto gridEW  = (unsigned)((E + 7) / 8);  // one wave per edge

    // normalization
    k_deg_init<<<gridN, B, 0, stream>>>(dinv, N);
    k_deg_accum<<<gridE, B, 0, stream>>>(dst, dinv, E);
    k_dinv<<<gridN, B, 0, stream>>>(dinv, N);

    // weight repack + input conversion
    k_repack_w<<<64, B, 0, stream>>>(W1, Wf1);
    k_repack_w<<<64, B, 0, stream>>>(W2, Wf2);
    k_cvt_bf16<<<gridNC, B, 0, stream>>>(x, Ab, N * CH);

    // layer 1
    k_gemm_bf16<<<gemmBlocks, B, 0, stream>>>(Ab, Wf1, Hbuf, N, mtiles);
    k_selfloop_init<<<gridNW, B, 0, stream>>>(Hbuf, dinv, OUT, N);
    k_edge_scatter<<<gridEW, B, 0, stream>>>(src, dst, dinv, Hbuf, OUT, E);
    k_bias_relu_bf16<<<gridNC, B, 0, stream>>>(OUT, b1, Ab, N);

    // layer 2
    k_gemm_bf16<<<gemmBlocks, B, 0, stream>>>(Ab, Wf2, Hbuf, N, mtiles);
    k_selfloop_init<<<gridNW, B, 0, stream>>>(Hbuf, dinv, OUT, N);
    k_edge_scatter<<<gridEW, B, 0, stream>>>(src, dst, dinv, Hbuf, OUT, E);
    k_bias_relu_f32<<<gridNC, B, 0, stream>>>(OUT, b2, N);

    // pooling + heads
    k_pool_zero<<<(unsigned)((G * CH + B - 1) / B), B, 0, stream>>>(gsum, gcnt, G);
    k_pool_cnt<<<gridN, B, 0, stream>>>(bat, gcnt, N);
    k_pool_sum<<<gridNW, B, 0, stream>>>(bat, OUT, gsum, N);
    k_gmean<<<(unsigned)((G * CH + B - 1) / B), B, 0, stream>>>(gsum, gcnt, G);
    k_head<<<(unsigned)((G * LAT + B - 1) / B), B, 0, stream>>>(gsum, Wmu, bmu, Wlv, blv, out, G);
}
